// DescLayer_61229053772255
// MI455X (gfx1250) — matrix operand order repaired and checked by reading, not compiled, not run on hardware
//
#include <hip/hip_runtime.h>
#include <hip/hip_bf16.h>

// Problem constants (B=2, S=512, IN=OUT=64, NB=8)
#define TOKENS   1024
#define NDIM     64
#define NB       8
#define KVALS    512
#define LN_EPS   1e-5f
#define TWO_PI_F 6.283185307179586f

typedef __attribute__((ext_vector_type(16))) _Float16 v16h;
typedef __attribute__((ext_vector_type(8)))  _Float16 v8h;
typedef __attribute__((ext_vector_type(8)))  float    v8f;

// ---------------------------------------------------------------------------
// Stage 0: pack fp16 combined weight [128 out x 64 in] = [M ; R] and
// precompute invper[e] = 2*pi / periods[e]  (k-independent, hoists all divides)
// ---------------------------------------------------------------------------
__global__ void __launch_bounds__(256)
k_prep(const float* __restrict__ M, const float* __restrict__ R,
       const float* __restrict__ periods,
       _Float16* __restrict__ wgtB, float* __restrict__ invper) {
  const int e = blockIdx.x * 256 + threadIdx.x;       // 0..32767
  invper[e] = TWO_PI_F / periods[e];
  if (e < 128 * NDIM) {
    const int o = e >> 6, j = e & 63;
    const float w = (o < NDIM) ? M[o * NDIM + j] : R[(o - NDIM) * NDIM + j];
    wgtB[e] = (_Float16)w;
  }
}

// ---------------------------------------------------------------------------
// Stage 1: LayerNorm, one wave32 per token, fp16 output for the WMMA GEMM
// ---------------------------------------------------------------------------
__global__ void __launch_bounds__(128)
k_layernorm(const float* __restrict__ x, const float* __restrict__ gamma,
            const float* __restrict__ beta, _Float16* __restrict__ lnh) {
  const int lane = threadIdx.x & 31;
  const int tok  = blockIdx.x * 4 + (threadIdx.x >> 5);
  const float* xr = x + (size_t)tok * NDIM;
  float v0 = xr[lane], v1 = xr[lane + 32];
  float s = v0 + v1;
#pragma unroll
  for (int off = 16; off; off >>= 1) s += __shfl_xor(s, off, 32);
  const float mu = s * (1.0f / 64.0f);
  const float d0 = v0 - mu, d1 = v1 - mu;
  float q = d0 * d0 + d1 * d1;
#pragma unroll
  for (int off = 16; off; off >>= 1) q += __shfl_xor(q, off, 32);
  const float rs = rsqrtf(q * (1.0f / 64.0f) + LN_EPS);
  _Float16* lr = lnh + (size_t)tok * NDIM;
  lr[lane]      = (_Float16)(d0 * rs * gamma[lane]      + beta[lane]);
  lr[lane + 32] = (_Float16)(d1 * rs * gamma[lane + 32] + beta[lane + 32]);
}

// ---------------------------------------------------------------------------
// Stage 2: projection GEMM via v_wmma_f32_16x16x32_f16
//   D[16 tok x 16 out] += A[16 tok x 32 k] * B[32 k x 16 out], 2 k-steps.
//   One tile per wave; 512 tiles = 128 blocks x 4 waves. No divergence
//   anywhere (EXEC all-ones for WMMA; stores are unconditional into a
//   fused [token][128] output where cols 0..63 = x_proj, 64..127 = res).
// A layout (ISA 7.12.2, 16-bit A 16x32): lane L<16 -> row L, halves 0..7 =
//   K 0..7, halves 8..15 = K 16..23; lane L>=16 -> row L-16, K+8.
// B layout (same k order as A): col = lane&15, halves 0..7 = K 0..7,
//   halves 8..15 = K 16..23; lanes 16..31 -> K+8.
// ---------------------------------------------------------------------------
__global__ void __launch_bounds__(128)
k_proj_wmma(const _Float16* __restrict__ lnh, const _Float16* __restrict__ wgtB,
            float* __restrict__ projres) {
  const int lane = threadIdx.x & 31;
  const int tile = blockIdx.x * 4 + (threadIdx.x >> 5);  // 0..511
  const int tokenTile = tile >> 3;                       // 64 token tiles
  const int outTile   = tile & 7;                        // 8 output tiles (128 outs)

  __builtin_prefetch(wgtB + (size_t)outTile * 16 * NDIM, 0, 0); // global_prefetch_b8

  const int arow = lane & 15;
  const int akb  = (lane >> 4) << 3;    // 0 or 8
  const int bcol = outTile * 16 + (lane & 15);
  const int bkb  = (lane >> 4) << 3;    // 0 or 8
  const _Float16* arowp = lnh  + (size_t)(tokenTile * 16 + arow) * NDIM;
  const _Float16* bcolp = wgtB + (size_t)bcol * NDIM;

  v8f acc = {};
#pragma unroll
  for (int c = 0; c < NDIM; c += 32) {
    union { v16h v; v8h h[2]; } A, Bm;
    A.h[0]  = *(const v8h*)(arowp + c + akb);        // K = c+akb .. +7
    A.h[1]  = *(const v8h*)(arowp + c + akb + 16);   // K = c+16+akb .. +7
    Bm.h[0] = *(const v8h*)(bcolp + c + bkb);        // K = c+bkb .. +7
    Bm.h[1] = *(const v8h*)(bcolp + c + bkb + 16);   // K = c+16+bkb .. +7
    acc = __builtin_amdgcn_wmma_f32_16x16x32_f16(
        /*neg_a=*/false, A.v, /*neg_b=*/false, Bm.v,
        /*c_mod=*/(short)0, acc, /*reuse_a=*/false, /*reuse_b=*/false);
  }

  // D layout: VGPR r -> M=r (lanes 0-15) / M=r+8 (lanes 16-31); N = lane&15.
  // Unconditional store into fused [token][128] row.
  const int Nc = outTile * 16 + (lane & 15);
  const int mh = (lane >> 4) << 3;
  float* base = projres + (size_t)(tokenTile * 16 + mh) * 128 + Nc;
#pragma unroll
  for (int r = 0; r < 8; ++r)
    base[(size_t)r * 128] = acc[r];
}

// ---------------------------------------------------------------------------
// Stage 3: W table, W[k][i][j] = sum_g P[i,j,g] * cos(k * invper[i,j,g]).
// 512 blocks (one per k) x 256 threads; 128 fast-cos per thread.
// ---------------------------------------------------------------------------
__global__ void __launch_bounds__(256)
k_wtable(const float* __restrict__ P, const float* __restrict__ invper,
         float* __restrict__ Wtab) {
  const int   kv = blockIdx.x;          // 0..511
  const float kf = (float)kv;
  float* Wk = Wtab + (size_t)kv * (NDIM * NDIM);
  for (int e = threadIdx.x; e < NDIM * NDIM; e += 256) {
    const float* Pe = P      + (size_t)e * NB;
    const float* Ie = invper + (size_t)e * NB;
    float acc = 0.0f;
#pragma unroll
    for (int g = 0; g < NB; ++g)
      acc += Pe[g] * __cosf(kf * Ie[g]);
    Wk[e] = acc;
  }
}

// ---------------------------------------------------------------------------
// Stage 4: out[t] = res[t] + W_{k[t]} @ xproj[t]; one wave32 per token,
// 2 output rows per lane, xproj broadcast via wave shuffles.
// projres row layout: [0..63] = x_proj, [64..127] = res.
// ---------------------------------------------------------------------------
__global__ void __launch_bounds__(128)
k_final(const int* __restrict__ karr, const float* __restrict__ projres,
        const float* __restrict__ Wtab, float* __restrict__ out) {
  const int lane = threadIdx.x & 31;
  const int tok  = blockIdx.x * 4 + (threadIdx.x >> 5);
  const int kv   = karr[tok] & (KVALS - 1);
  const float* Wk = Wtab    + (size_t)kv  * (NDIM * NDIM);
  const float* pr = projres + (size_t)tok * 128;
  const float x0 = pr[lane], x1 = pr[lane + 32];
  const float* w0 = Wk + (size_t)lane * NDIM;
  const float* w1 = Wk + (size_t)(lane + 32) * NDIM;
  float a0 = 0.0f, a1 = 0.0f;
#pragma unroll 8
  for (int j = 0; j < 32; ++j) {
    const float xj = __shfl(x0, j, 32);
    a0 += w0[j] * xj;  a1 += w1[j] * xj;
  }
#pragma unroll 8
  for (int j = 0; j < 32; ++j) {
    const float xj = __shfl(x1, j, 32);
    a0 += w0[j + 32] * xj;  a1 += w1[j + 32] * xj;
  }
  out[(size_t)tok * NDIM + lane]      = pr[64 + lane] + a0;
  out[(size_t)tok * NDIM + lane + 32] = pr[96 + lane] + a1;
}

// ---------------------------------------------------------------------------
extern "C" void kernel_launch(void* const* d_in, const int* in_sizes, int n_in,
                              void* d_out, int out_size, void* d_ws, size_t ws_size,
                              hipStream_t stream) {
  (void)in_sizes; (void)n_in; (void)out_size; (void)ws_size;
  const float* x       = (const float*)d_in[0];
  const int*   karr    = (const int*)  d_in[1];   // int32 per harness convention
  const float* M       = (const float*)d_in[2];
  const float* R       = (const float*)d_in[3];
  const float* P       = (const float*)d_in[4];
  const float* gamma   = (const float*)d_in[5];
  const float* beta    = (const float*)d_in[6];
  const float* periods = (const float*)d_in[7];
  float* out = (float*)d_out;

  // Workspace layout (all offsets 128B-aligned), total ~8.8 MB
  char* ws = (char*)d_ws;
  float*    Wtab    = (float*)   (ws + 0);         //  512*64*64 f32 = 8 MB
  _Float16* lnh     = (_Float16*)(ws + 8388608);   // 1024*64 f16
  _Float16* wgtB    = (_Float16*)(ws + 8519680);   //  128*64 f16
  float*    invper  = (float*)   (ws + 8536064);   // 32768 f32
  float*    projres = (float*)   (ws + 8667136);   // 1024*128 f32 (xproj|res)

  k_prep     <<<128, 256, 0, stream>>>(M, R, periods, wgtB, invper);
  k_layernorm<<<256, 128, 0, stream>>>(x, gamma, beta, lnh);
  k_proj_wmma<<<128, 128, 0, stream>>>(lnh, wgtB, projres);
  k_wtable   <<<512, 256, 0, stream>>>(P, invper, Wtab);
  k_final    <<<256, 128, 0, stream>>>(karr, projres, Wtab, out);
}
